// FastGRU_3478923509773
// MI455X (gfx1250) — compile-verified, matrix operand order checked by reading, not run on hardware
//
#include <hip/hip_runtime.h>
#include <hip/hip_bf16.h>

// Shapes fixed by the reference: B=4, L=4096, d=256, w=8, H=512 (pad=0).
// P = B*L = 16384 positions, C = 256 channels.

#define NPOS 16384
#define NCH  256
#define NSLICE (NPOS * NCH)            // 4,194,304 elements per (dir,gate) slice
#define WELEMS (NCH * NCH)             // 65,536 elements per weight matrix

typedef __attribute__((ext_vector_type(16))) __bf16 v16bf;
typedef __attribute__((ext_vector_type(8)))  float  v8f;

// ---------- helpers ----------

static __device__ __forceinline__ unsigned short f2bf(float f) {
  union { float f; unsigned u; } v; v.f = f;
  unsigned u = v.u;
  u += 0x7fffu + ((u >> 16) & 1u);     // round-to-nearest-even
  return (unsigned short)(u >> 16);
}

static __device__ __forceinline__ float sigmoidf_(float x) {
  return 1.0f / (1.0f + __expf(-x));
}

// Load one 16x16x32 bf16 WMMA A/B fragment from a row-major [*][256] bf16
// matrix. Lane layout (ISA 7.12.2, 16-bit A 16x32; B mirrors with N<->M):
//   lanes 0-15 : row = lane,    K = kk*32 + {0..7} and {16..23}
//   lanes 16-31: row = lane-16, K = kk*32 + {8..15} and {24..31}
// i.e. two contiguous 16-byte loads at k1 = kk*32 + hi*8 and k1+16.
static __device__ __forceinline__ v16bf load_frag(const unsigned short* base,
                                                  int row, int kk, int hi) {
  const int k1 = kk * 32 + hi * 8;
  const unsigned short* p = base + row * NCH + k1;
  v16bf a;
  uint4* d = reinterpret_cast<uint4*>(&a);
  d[0] = *reinterpret_cast<const uint4*>(p);
  d[1] = *reinterpret_cast<const uint4*>(p + 16);
  return a;
}

static __device__ __forceinline__ v8f wmma_bf16(v16bf a, v16bf b, v8f c) {
  return __builtin_amdgcn_wmma_f32_16x16x32_bf16(false, a, false, b,
                                                 (short)0, c, false, false);
}

// ---------- kernel 1: x (f32) -> bf16 ----------

__global__ void fg_cvt_x(const float* __restrict__ x,
                         unsigned short* __restrict__ xbf, int n) {
  int g = blockIdx.x * blockDim.x + threadIdx.x;
  if (g < n) xbf[g] = f2bf(x[g]);
}

// ---------- kernel 2: transpose + convert weights ----------
// WxT/WhT layout: [dir*3+gate][n][k] bf16, gate order z,r,h.

__global__ void fg_cvt_w(const float* __restrict__ Wxz, const float* __restrict__ Whz,
                         const float* __restrict__ Wxr, const float* __restrict__ Whr,
                         const float* __restrict__ Wxh, const float* __restrict__ Whh,
                         unsigned short* __restrict__ WxT,
                         unsigned short* __restrict__ WhT) {
  int g = blockIdx.x * blockDim.x + threadIdx.x;   // < 12 * 65536
  if (g >= 12 * WELEMS) return;
  int dg = g >> 16;                 // dir*3 + gate
  int r  = g & 0xFFFF;
  int n  = r >> 8;
  int k  = r & 0xFF;
  int dir = dg / 3, gate = dg % 3;
  const float* Wx = (gate == 0) ? Wxz : (gate == 1) ? Wxr : Wxh;
  const float* Wh = (gate == 0) ? Whz : (gate == 1) ? Whr : Whh;
  size_t src = (size_t)dir * WELEMS + (size_t)k * NCH + n;  // W[dir][k][n]
  WxT[g] = f2bf(Wx[src]);
  WhT[g] = f2bf(Wh[src]);
}

// ---------- kernel 3: input projections  U[dg] = X @ Wx[dg] + bias ----------
// grid: (1024 M-tiles, 12 dir*gate), block 512 (16 waves, wave owns 16 cols).

__global__ void fg_proj(const unsigned short* __restrict__ Xbf,
                        const unsigned short* __restrict__ WxT,
                        const float* __restrict__ bz, const float* __restrict__ br,
                        const float* __restrict__ bh,
                        float* __restrict__ U) {
  const int dg   = blockIdx.y;
  const int dir  = dg / 3, gate = dg % 3;
  const int wave = threadIdx.x >> 5;
  const int lane = threadIdx.x & 31;
  const int lr   = lane & 15;
  const int hi   = lane >> 4;
  const int n0   = wave * 16;
  const int col  = n0 + lr;
  const int m0   = blockIdx.x * 16;

  const float* bias = ((gate == 0) ? bz : (gate == 1) ? br : bh) + dir * NCH;
  const float bv = bias[col];
  v8f acc;
#pragma unroll
  for (int j = 0; j < 8; ++j) acc[j] = bv;

  const unsigned short* A = Xbf + (size_t)m0 * NCH;
  const unsigned short* B = WxT + (size_t)dg * WELEMS;
#pragma unroll
  for (int kk = 0; kk < 8; ++kk) {
    v16bf a = load_frag(A, lr, kk, hi);
    v16bf b = load_frag(B, n0 + lr, kk, hi);
    acc = wmma_bf16(a, b, acc);
  }

  float* Up = U + (size_t)dg * NSLICE;
#pragma unroll
  for (int j = 0; j < 8; ++j)
    Up[(size_t)(m0 + j + hi * 8) * NCH + col] = acc[j];
}

// ---------- kernel 4: fused GRU recurrence ----------
// 260 blocks: [0,128) dir0 (w-scan fwd), [128,256) dir1 (w-scan rev),
// {256,257} dir2 (H-scan fwd), {258,259} dir3 (H-scan rev).
// Each block: 16-row tile, h in LDS (bf16 A-operand), 16 waves x 16 cols.
// Wh B-fragments for the z and r gates are hoisted into registers (loop
// invariant, 128 VGPRs) so the serial critical path of each step touches
// only LDS + registers; the n-gate weights stream from L2 inside the
// pipelined phase-2 loop. Output h_t overwrites the consumed U_z slot
// (read-before-write by the same lane).

__global__ void fg_gru_rec(const unsigned short* __restrict__ WhT,
                           float* __restrict__ U) {
  __shared__ __align__(16) unsigned short hbuf[16 * NCH];
  __shared__ __align__(16) unsigned short rhbuf[16 * NCH];

  int bx = blockIdx.x, dir, mtile;
  if (bx < 128)       { dir = 0; mtile = bx; }
  else if (bx < 256)  { dir = 1; mtile = bx - 128; }
  else if (bx < 258)  { dir = 2; mtile = bx - 256; }
  else                { dir = 3; mtile = bx - 258; }
  const int T = (dir < 2) ? 8 : 512;

  const int tid  = threadIdx.x;
  const int wave = tid >> 5;
  const int lane = tid & 31;
  const int lr   = lane & 15;
  const int hi   = lane >> 4;
  const int n0   = wave * 16;
  const int col  = n0 + lr;
  const int m0   = mtile * 16;

  const unsigned short* Bz = WhT + (size_t)(dir * 3 + 0) * WELEMS;
  const unsigned short* Br = WhT + (size_t)(dir * 3 + 1) * WELEMS;
  const unsigned short* Bh = WhT + (size_t)(dir * 3 + 2) * WELEMS;
  float* Uz = U + (size_t)(dir * 3 + 0) * NSLICE;   // also receives output h
  float* Ur = U + (size_t)(dir * 3 + 1) * NSLICE;
  float* Uh = U + (size_t)(dir * 3 + 2) * NSLICE;

  // Loop-invariant B fragments for the z/r gates -> registers (8+8 frags).
  v16bf Bzf[8], Brf[8];
#pragma unroll
  for (int kk = 0; kk < 8; ++kk) {
    Bzf[kk] = load_frag(Bz, n0 + lr, kk, hi);
    Brf[kk] = load_frag(Br, n0 + lr, kk, hi);
  }

  for (int i = tid; i < 16 * NCH; i += 512) hbuf[i] = 0;
  __syncthreads();

  float h[8];
#pragma unroll
  for (int j = 0; j < 8; ++j) h[j] = 0.0f;

#pragma unroll 1
  for (int t = 0; t < T; ++t) {
    const int tp = (dir & 1) ? (T - 1 - t) : t;
    int fl[8];
#pragma unroll
    for (int j = 0; j < 8; ++j) {
      int r = m0 + j + hi * 8;
      // dir 0/1: rows are (b,H) pairs, scan over w  -> flat = r*8 + tp
      // dir 2/3: rows are (b,w) pairs, scan over H  -> flat = b*4096 + tp*8 + ww
      fl[j] = (dir < 2) ? (r * 8 + tp)
                        : (((r >> 3) * 4096) + tp * 8 + (r & 7));
    }

    v8f accz, accr;
#pragma unroll
    for (int j = 0; j < 8; ++j) {
      accz[j] = Uz[(size_t)fl[j] * NCH + col];
      accr[j] = Ur[(size_t)fl[j] * NCH + col];
    }
#pragma unroll
    for (int kk = 0; kk < 8; ++kk) {
      v16bf a = load_frag(hbuf, lr, kk, hi);   // shared A fragment (LDS)
      accz = wmma_bf16(a, Bzf[kk], accz);
      accr = wmma_bf16(a, Brf[kk], accr);
    }

    float zv[8], rv[8];
#pragma unroll
    for (int j = 0; j < 8; ++j) {
      zv[j] = sigmoidf_(accz[j]);
      rv[j] = sigmoidf_(accr[j]);
      rhbuf[(j + hi * 8) * NCH + col] = f2bf(rv[j] * h[j]);
    }
    __syncthreads();   // rh visible to all waves

    v8f accn;
#pragma unroll
    for (int j = 0; j < 8; ++j) accn[j] = Uh[(size_t)fl[j] * NCH + col];
#pragma unroll
    for (int kk = 0; kk < 8; ++kk) {
      v16bf a = load_frag(rhbuf, lr, kk, hi);
      v16bf b = load_frag(Bh, n0 + lr, kk, hi);   // L2-hot, pipelined
      accn = wmma_bf16(a, b, accn);
    }

#pragma unroll
    for (int j = 0; j < 8; ++j) {
      float nv = tanhf(accn[j]);
      float hn = (1.0f - zv[j]) * h[j] + zv[j] * nv;
      h[j] = hn;
      Uz[(size_t)fl[j] * NCH + col] = hn;            // emit h_t (reuses U_z)
      hbuf[(j + hi * 8) * NCH + col] = f2bf(hn);     // next step's A operand
    }
    __syncthreads();   // h update visible before next step's z/r matmuls
  }
}

// ---------- kernel 5: sum the four directions (vectorized) ----------

__global__ void fg_sum(const float* __restrict__ U, float* __restrict__ out, int n4) {
  int g = blockIdx.x * blockDim.x + threadIdx.x;
  if (g >= n4) return;
  const float4* u0 = (const float4*)(U + (size_t)0 * 3 * NSLICE);
  const float4* u1 = (const float4*)(U + (size_t)1 * 3 * NSLICE);
  const float4* u2 = (const float4*)(U + (size_t)2 * 3 * NSLICE);
  const float4* u3 = (const float4*)(U + (size_t)3 * 3 * NSLICE);
  float4 a = u0[g], b = u1[g], c = u2[g], d = u3[g];
  float4 r;
  r.x = a.x + b.x + c.x + d.x;
  r.y = a.y + b.y + c.y + d.y;
  r.z = a.z + b.z + c.z + d.z;
  r.w = a.w + b.w + c.w + d.w;
  ((float4*)out)[g] = r;
}

// ---------- launch ----------

extern "C" void kernel_launch(void* const* d_in, const int* in_sizes, int n_in,
                              void* d_out, int out_size, void* d_ws, size_t ws_size,
                              hipStream_t stream) {
  const float* x   = (const float*)d_in[0];
  const float* Wxz = (const float*)d_in[1];
  const float* Whz = (const float*)d_in[2];
  const float* Wxr = (const float*)d_in[3];
  const float* Whr = (const float*)d_in[4];
  const float* Wxh = (const float*)d_in[5];
  const float* Whh = (const float*)d_in[6];
  const float* bz  = (const float*)d_in[7];
  const float* br  = (const float*)d_in[8];
  const float* bh  = (const float*)d_in[9];
  float* out = (float*)d_out;

  // workspace layout (bytes)
  char* ws = (char*)d_ws;
  unsigned short* Xbf = (unsigned short*)(ws);                         //  8.4 MB
  unsigned short* WxT = (unsigned short*)(ws + 8388608);               //  1.5 MB
  unsigned short* WhT = (unsigned short*)(ws + 9961472);               //  1.5 MB
  float*          U   = (float*)(ws + 11534336);                       // 201.3 MB
  (void)in_sizes; (void)n_in; (void)out_size; (void)ws_size;

  fg_cvt_x<<<(NSLICE + 255) / 256, 256, 0, stream>>>(x, Xbf, NSLICE);
  fg_cvt_w<<<(12 * WELEMS + 255) / 256, 256, 0, stream>>>(
      Wxz, Whz, Wxr, Whr, Wxh, Whh, WxT, WhT);
  fg_proj<<<dim3(NPOS / 16, 12), 512, 0, stream>>>(Xbf, WxT, bz, br, bh, U);
  fg_gru_rec<<<260, 512, 0, stream>>>(WhT, U);
  fg_sum<<<(NSLICE / 4 + 255) / 256, 256, 0, stream>>>(U, out, NSLICE / 4);
}
